// PartAttention_25434796327222
// MI455X (gfx1250) — matrix operand order repaired and checked by reading, not compiled, not run on hardware
//
#include <hip/hip_runtime.h>

typedef __attribute__((ext_vector_type(16))) _Float16 v16h;
typedef __attribute__((ext_vector_type(8)))  float    v8f;

struct alignas(16) f16x8 { _Float16 h[8]; };
union AV { v16h v; f16x8 q[2]; };

#define NB   1024
#define NT   16
#define NJ   17
#define NE   256
#define NEH  128

__constant__ int LIMB_TAB[4][3] = {{5,7,9},{6,8,10},{11,13,15},{12,14,16}};

// ---------------- K0: convert + transpose weights to f16 [n][k] layout ----------------
__global__ void k_prep(const float* __restrict__ wdec_in, const float* __restrict__ whead_in,
                       const float* __restrict__ wlimb_in,
                       _Float16* __restrict__ wdec, _Float16* __restrict__ whead,
                       _Float16* __restrict__ wlimb) {
  long i = (long)blockIdx.x * blockDim.x + threadIdx.x;
  if (i < 128l * 256) {                     // proj_dec_w (256,128) -> [n][k]
    int n = (int)(i >> 8), k = (int)(i & 255);
    wdec[i] = (_Float16)wdec_in[k * 128 + n];
    return;
  }
  i -= 128l * 256;
  if (i < 15l * 128 * 256) {                // head_conv_w (3,5,256,128) -> [kh*5+kw][n][c]
    long kk = i >> 15;
    int n = (int)((i >> 8) & 127), c = (int)(i & 255);
    whead[i] = (_Float16)whead_in[(kk * 256 + c) * 128 + n];
    return;
  }
  i -= 15l * 128 * 256;
  if (i < 36l * 128 * 256) {                // limb_conv_w (4,3,3,256,128) -> [p*9+kh*3+kw][n][c]
    long kk = i >> 15;
    int n = (int)((i >> 8) & 127), c = (int)(i & 255);
    wlimb[i] = (_Float16)wlimb_in[(kk * 256 + c) * 128 + n];
  }
}

// ---------------- K1: full = pose @ proj_in_w + b  (K=3, VALU), stored f16 ----------------
__global__ void k_proj_in(const float* __restrict__ pose, const float* __restrict__ w,
                          const float* __restrict__ b, _Float16* __restrict__ full) {
  long i = (long)blockIdx.x * blockDim.x + threadIdx.x;   // one elem per thread
  if (i >= (long)NB * NT * NJ * NE) return;
  int c = (int)(i & 255);
  long r = i >> 8;
  const float* p = pose + r * 3;
  float acc = b[c] + p[0] * w[c] + p[1] * w[256 + c] + p[2] * w[512 + c];
  full[i] = (_Float16)acc;
}

// ---------------- WMMA fragment helpers ----------------
__device__ inline v16h load_a_frag(const _Float16* rowp, int k0, int half) {
  // A lane (row fixed): K = {k0+8h .. k0+8h+7, k0+16+8h .. k0+16+8h+7}
  AV a;
  a.q[0] = *(const f16x8*)(rowp + k0 + 8 * half);
  a.q[1] = *(const f16x8*)(rowp + k0 + 16 + 8 * half);
  return a.v;
}
__device__ inline v16h zero_frag() {
  AV a;
  #pragma unroll
  for (int z = 0; z < 8; ++z) { a.q[0].h[z] = (_Float16)0; a.q[1].h[z] = (_Float16)0; }
  return a.v;
}
__device__ inline v16h load_b_frag(const _Float16* colp, int k0, int half) {
  // B lane (col fixed): K = {k0+8h .. k0+8h+7, k0+16+8h .. k0+16+8h+7}
  AV b;
  b.q[0] = *(const f16x8*)(colp + k0 + 8 * half);
  b.q[1] = *(const f16x8*)(colp + k0 + 16 + 8 * half);
  return b.v;
}

// ---------------- K2: fullp = full @ proj_dec_w + b  (M=278528, K=256, N=128) ----------------
__global__ void __launch_bounds__(256) k_gemm_dec(const _Float16* __restrict__ full,
                                                  const _Float16* __restrict__ wdec,
                                                  const float* __restrict__ bias,
                                                  _Float16* __restrict__ fullp) {
  int lane = threadIdx.x & 31;
  int wv   = threadIdx.x >> 5;            // 8 waves -> 8 col tiles
  int m0   = blockIdx.x * 16;
  int half = lane >> 4;
  int row  = m0 + (lane & 15);
  int n    = wv * 16 + (lane & 15);
  const _Float16* ap = full + (long)row * 256;
  const _Float16* bp = wdec + (long)n * 256;
  v8f acc = {};
  #pragma unroll
  for (int kb = 0; kb < 8; ++kb) {
    v16h a = load_a_frag(ap, kb * 32, half);
    v16h b = load_b_frag(bp, kb * 32, half);
    acc = __builtin_amdgcn_wmma_f32_16x16x32_f16(false, a, false, b, (short)0, acc, false, false);
  }
  float bn = bias[n];
  #pragma unroll
  for (int v = 0; v < 8; ++v) {
    int rr = m0 + v + 8 * half;
    fullp[(long)rr * 128 + n] = (_Float16)(acc[v] + bn);
  }
}

// ---------------- K3: head conv (3x5 SAME over T x 5joints) as implicit GEMM ----------------
__global__ void __launch_bounds__(256) k_conv_head(const _Float16* __restrict__ full,
                                                   const _Float16* __restrict__ whead,
                                                   const float* __restrict__ bias,
                                                   _Float16* __restrict__ ench) {
  int lane = threadIdx.x & 31;
  int wv   = threadIdx.x >> 5;
  int m0   = blockIdx.x * 16;
  int half = lane >> 4;
  int row  = m0 + (lane & 15);            // flattened (b,t,j), j in 0..4
  int n    = wv * 16 + (lane & 15);
  int j  = row % 5;
  int bt = row / 5;
  int t  = bt & 15;
  int b  = bt >> 4;
  v8f acc = {};
  for (int kh = 0; kh < 3; ++kh) {
    int tt = t + kh - 1;
    bool vt = (unsigned)tt < 16u;
    for (int kw = 0; kw < 5; ++kw) {
      int jj = j + kw - 2;
      bool valid = vt && ((unsigned)jj < 5u);
      const _Float16* ap = full + (((long)b * 16 + (valid ? tt : 0)) * 17 + (valid ? jj : 0)) * 256;
      const _Float16* bp = whead + ((long)(kh * 5 + kw) * 128 + n) * 256;
      #pragma unroll
      for (int kb = 0; kb < 8; ++kb) {
        v16h a = valid ? load_a_frag(ap, kb * 32, half) : zero_frag();
        v16h bm = load_b_frag(bp, kb * 32, half);
        acc = __builtin_amdgcn_wmma_f32_16x16x32_f16(false, a, false, bm, (short)0, acc, false, false);
      }
    }
  }
  float bn = bias[n];
  #pragma unroll
  for (int v = 0; v < 8; ++v) {
    int rr = m0 + v + 8 * half;
    ench[(long)rr * 128 + n] = (_Float16)(acc[v] + bn);
  }
}

// ---------------- K4: limb convs (3x3 SAME over T x 3joints, 4 limbs) ----------------
__global__ void __launch_bounds__(256) k_conv_limb(const _Float16* __restrict__ full,
                                                   const _Float16* __restrict__ wlimb,
                                                   const float* __restrict__ bias,
                                                   _Float16* __restrict__ encl) {
  int lane = threadIdx.x & 31;
  int wv   = threadIdx.x >> 5;
  int p    = blockIdx.x / 3072;           // 49152/16 row-tiles per limb
  int m0   = (blockIdx.x % 3072) * 16;
  int half = lane >> 4;
  int row  = m0 + (lane & 15);            // flattened (b,t,j) within limb, j in 0..2
  int n    = wv * 16 + (lane & 15);
  int j  = row % 3;
  int bt = row / 3;
  int t  = bt & 15;
  int b  = bt >> 4;
  v8f acc = {};
  for (int kh = 0; kh < 3; ++kh) {
    int tt = t + kh - 1;
    bool vt = (unsigned)tt < 16u;
    for (int kw = 0; kw < 3; ++kw) {
      int jj = j + kw - 1;
      bool valid = vt && ((unsigned)jj < 3u);
      int joint = valid ? LIMB_TAB[p][jj] : 0;
      const _Float16* ap = full + (((long)b * 16 + (valid ? tt : 0)) * 17 + joint) * 256;
      const _Float16* bp = wlimb + ((long)((p * 9 + kh * 3 + kw) * 128 + n)) * 256;
      #pragma unroll
      for (int kb = 0; kb < 8; ++kb) {
        v16h a = valid ? load_a_frag(ap, kb * 32, half) : zero_frag();
        v16h bm = load_b_frag(bp, kb * 32, half);
        acc = __builtin_amdgcn_wmma_f32_16x16x32_f16(false, a, false, bm, (short)0, acc, false, false);
      }
    }
  }
  float bn = bias[p * 128 + n];
  #pragma unroll
  for (int v = 0; v < 8; ++v) {
    int rr = m0 + v + 8 * half;
    encl[((long)p * 49152 + rr) * 128 + n] = (_Float16)(acc[v] + bn);
  }
}

// ---------------- K5: dec mix + layernorm + 128->3 out-proj + residual scatter ----------------
__global__ void __launch_bounds__(128) k_final(const float* __restrict__ pose,
    const _Float16* __restrict__ fullp, const _Float16* __restrict__ ench,
    const _Float16* __restrict__ encl,
    const float* __restrict__ hd_w, const float* __restrict__ hd_b,
    const float* __restrict__ h_g,  const float* __restrict__ h_bb,
    const float* __restrict__ h_ow, const float* __restrict__ h_ob,
    const float* __restrict__ ld_w, const float* __restrict__ ld_b,
    const float* __restrict__ l_g,  const float* __restrict__ l_bb,
    const float* __restrict__ l_ow, const float* __restrict__ l_ob,
    float* __restrict__ out) {
  long blk = blockIdx.x;                  // B*T*17 rows
  int  o   = (int)(blk % 17);
  long bt  = blk / 17;
  int  d   = threadIdx.x;                 // channel 0..127
  __shared__ float red[128];

  float dec;
  int joint;
  const float *g, *bb, *ow, *ob;
  if (o < 5) {                            // head row
    joint = o;
    float s = hd_b[o];
    #pragma unroll
    for (int i = 0; i < 10; ++i) {
      float cat = (i < 5) ? (float)fullp[(bt * 17 + i) * 128 + d]
                          : (float)ench[(bt * 5 + (i - 5)) * 128 + d];
      s += hd_w[o * 10 + i] * cat;
    }
    dec = s; g = h_g; bb = h_bb; ow = h_ow; ob = h_ob;
  } else {                                // limb row
    int k = o - 5, p = k / 3, oo = k % 3;
    joint = LIMB_TAB[p][oo];
    float s = ld_b[p * 3 + oo];
    #pragma unroll
    for (int i = 0; i < 6; ++i) {
      float cat = (i < 3) ? (float)fullp[(bt * 17 + LIMB_TAB[p][i]) * 128 + d]
                          : (float)encl[((long)p * 49152 + bt * 3 + (i - 3)) * 128 + d];
      s += ld_w[(p * 3 + oo) * 6 + i] * cat;
    }
    dec = s; g = l_g + p * 128; bb = l_bb + p * 128; ow = l_ow + (long)p * 128 * 3; ob = l_ob + p * 3;
  }

  auto block_sum = [&](float v) -> float {
    red[d] = v; __syncthreads();
    #pragma unroll
    for (int s = 64; s > 0; s >>= 1) { if (d < s) red[d] += red[d + s]; __syncthreads(); }
    float r = red[0]; __syncthreads();
    return r;
  };

  float mean = block_sum(dec) * (1.0f / 128.0f);
  float xm = dec - mean;
  float var = block_sum(xm * xm) * (1.0f / 128.0f);
  float nrm = xm * rsqrtf(var + 1e-5f) * g[d] + bb[d];

  float c0 = block_sum(nrm * ow[d * 3 + 0]);
  float c1 = block_sum(nrm * ow[d * 3 + 1]);
  float c2 = block_sum(nrm * ow[d * 3 + 2]);

  if (d < 3) {
    float delta = (d == 0 ? c0 : (d == 1 ? c1 : c2)) + ob[d];
    long idx = (bt * 17 + joint) * 3 + d;
    out[idx] = pose[idx] + delta;
  }
}

extern "C" void kernel_launch(void* const* d_in, const int* in_sizes, int n_in,
                              void* d_out, int out_size, void* d_ws, size_t ws_size,
                              hipStream_t stream) {
  const float* pose        = (const float*)d_in[0];
  const float* proj_in_w   = (const float*)d_in[1];
  const float* proj_in_b   = (const float*)d_in[2];
  const float* proj_dec_w  = (const float*)d_in[3];
  const float* proj_dec_b  = (const float*)d_in[4];
  const float* head_conv_w = (const float*)d_in[5];
  const float* head_conv_b = (const float*)d_in[6];
  const float* head_dec_w  = (const float*)d_in[7];
  const float* head_dec_b  = (const float*)d_in[8];
  const float* head_ln_g   = (const float*)d_in[9];
  const float* head_ln_b   = (const float*)d_in[10];
  const float* head_out_w  = (const float*)d_in[11];
  const float* head_out_b  = (const float*)d_in[12];
  const float* limb_conv_w = (const float*)d_in[13];
  const float* limb_conv_b = (const float*)d_in[14];
  const float* limb_dec_w  = (const float*)d_in[15];
  const float* limb_dec_b  = (const float*)d_in[16];
  const float* limb_ln_g   = (const float*)d_in[17];
  const float* limb_ln_b   = (const float*)d_in[18];
  const float* limb_out_w  = (const float*)d_in[19];
  const float* limb_out_b  = (const float*)d_in[20];

  // f16 workspace layout (element offsets); total = 144,310,272 f16 = ~289 MB
  _Float16* ws    = (_Float16*)d_ws;
  _Float16* full  = ws;                       // B*T*17*256 = 71,303,168
  _Float16* fullp = ws + 71303168l;           // B*T*17*128 = 35,651,584
  _Float16* ench  = ws + 106954752l;          // B*T*5*128  = 10,485,760
  _Float16* encl  = ws + 117440512l;          // 4*B*T*3*128= 25,165,824
  _Float16* wdec  = ws + 142606336l;          // 128*256
  _Float16* whead = ws + 142639104l;          // 15*128*256
  _Float16* wlimb = ws + 143130624l;          // 36*128*256

  k_prep<<<6656, 256, 0, stream>>>(proj_dec_w, head_conv_w, limb_conv_w, wdec, whead, wlimb);
  k_proj_in<<<278528, 256, 0, stream>>>(pose, proj_in_w, proj_in_b, full);
  k_gemm_dec<<<17408, 256, 0, stream>>>(full, wdec, proj_dec_b, fullp);
  k_conv_head<<<5120, 256, 0, stream>>>(full, whead, head_conv_b, ench);
  k_conv_limb<<<12288, 256, 0, stream>>>(full, wlimb, limb_conv_b, encl);
  k_final<<<278528, 128, 0, stream>>>(pose, fullp, ench, encl,
      head_dec_w, head_dec_b, head_ln_g, head_ln_b, head_out_w, head_out_b,
      limb_dec_w, limb_dec_b, limb_ln_g, limb_ln_b, limb_out_w, limb_out_b,
      (float*)d_out);
}
